// SpliceGraph_3796751090385
// MI455X (gfx1250) — matrix operand order repaired and checked by reading, not compiled, not run on hardware
//
#include <hip/hip_runtime.h>

// ---------------------------------------------------------------------------
// GCN layer for MI455X (gfx1250, wave32, WMMA).
//   h  = x @ gcn_w                      (WMMA f16 -> f32, stored f16)
//   z  = scatter_add(h[src]*norm, dst) + h*dinv^2 + gcn_b ; z = tanh(z)
//   g  = sigmoid(z @ gate_w + gate_b)   (WMMA, fused epilogue)
//   out= BN(relu((1-g)*xl + g*z)),  xl = x @ lin_w + lin_b
//
// Weights are pre-swizzled once per launch into fragment-ordered fp16 so the
// inner WMMA loop loads each B fragment as one contiguous 32B v16h read.
// Epilogues use a full-tile fast path (no bounds checks) for interior blocks.
// ---------------------------------------------------------------------------

typedef __attribute__((ext_vector_type(16))) _Float16 v16h;
typedef __attribute__((ext_vector_type(8)))  float    v8f;

#define D 128              // feature dim (D_IN == D_HID == 128)
#define MT 64              // rows per block in GEMM kernels
#define GEMM_THREADS 128   // 4 waves of 32
#define NFRAG (8 * 4 * 32) // (j,kk,lane) fragments per weight matrix

// ---------------- init: deg=1 (self loop), zacc=0, BN partials=0 ------------
__global__ void k_init(float* __restrict__ deg, float* __restrict__ zacc,
                       float* __restrict__ colsum, float* __restrict__ colsq,
                       int N, int totalZ) {
  int i = blockIdx.x * blockDim.x + threadIdx.x;
  if (i < totalZ) zacc[i] = 0.0f;
  if (i < N) deg[i] = 1.0f;                 // self-loop contribution
  if (i < D) { colsum[i] = 0.0f; colsq[i] = 0.0f; }
}

// ------- pre-swizzle a 128x128 f32 weight into WMMA B-fragment order --------
// o[((j*4+kk)*32 + lane)*16 + e] = (f16) w[(kk*32 + (e<8?0:16) + (lane>>4)*8 + (e&7))*D + j*16 + (lane&15)]
__global__ void k_swizzle_w(const float* __restrict__ w, _Float16* __restrict__ o) {
  int jkk  = blockIdx.x;          // 0..31 = j*4 + kk
  int lane = threadIdx.x;         // 0..31
  int j = jkk >> 2, kk = jkk & 3;
  int col = j * 16 + (lane & 15);
  int half_id = lane >> 4;
  _Float16* op = o + ((size_t)jkk * 32 + lane) * 16;
#pragma unroll
  for (int e = 0; e < 16; ++e) {
    int krow = kk * 32 + ((e < 8) ? 0 : 16) + half_id * 8 + (e & 7);
    op[e] = (_Float16)w[(size_t)krow * D + col];
  }
}

// ---------------- degree: deg[dst]++ per edge -------------------------------
__global__ void k_degree(const int* __restrict__ dst, float* __restrict__ deg,
                         int E) {
  int e = blockIdx.x * blockDim.x + threadIdx.x;
  if (e < E) atomicAdd(&deg[dst[e]], 1.0f);
}

// ---------------- dinv = rsqrt(deg) (in place) ------------------------------
__global__ void k_dinv(float* __restrict__ deg, int N) {
  int i = blockIdx.x * blockDim.x + threadIdx.x;
  if (i < N) deg[i] = rsqrtf(fmaxf(deg[i], 1e-12f));
}

// ---------------- fused WMMA GEMM: h = x@gcn_w (f16), xl = x@lin_w + b ------
__global__ __launch_bounds__(GEMM_THREADS)
void k_gemm_x(const float* __restrict__ x,
              const v16h* __restrict__ gswz,   // gcn_w swizzled fragments
              const v16h* __restrict__ lswz,   // lin_w swizzled fragments
              const float* __restrict__ lin_b,
              _Float16* __restrict__ h_out, float* __restrict__ xl_out, int N) {
  const int tid  = threadIdx.x;
  const int wave = tid >> 5;
  const int lane = tid & 31;
  const int row0 = blockIdx.x * MT;
  const bool full = (row0 + MT) <= N;      // interior block: no bounds checks

  __shared__ _Float16 sA[MT][D + 8];
  // cooperative load + f32->f16 convert of the 64x128 A tile
  for (int idx = tid; idx < MT * D; idx += GEMM_THREADS) {
    int r = idx >> 7, c = idx & (D - 1);
    int gr = row0 + r;
    float v = (full || gr < N) ? x[(size_t)gr * D + c] : 0.0f;
    sA[r][c] = (_Float16)v;
  }
  __syncthreads();

  const int half_id = lane >> 4;   // 0: lanes 0-15, 1: lanes 16-31
  const int mloc    = lane & 15;

  // A fragments for this wave's 16-row tile, 4 k-steps of 32 (ISA 7.12.2)
  v16h a[4];
#pragma unroll
  for (int kk = 0; kk < 4; ++kk) {
#pragma unroll
    for (int e = 0; e < 16; ++e) {
      int kl = ((e < 8) ? 0 : 16) + half_id * 8 + (e & 7);
      a[kk][e] = sA[wave * 16 + mloc][kk * 32 + kl];
    }
  }

#pragma unroll
  for (int j = 0; j < 8; ++j) {            // 8 column tiles of 16
    v8f acc_h = {}; v8f acc_l = {};
    const int col = j * 16 + mloc;
#pragma unroll
    for (int kk = 0; kk < 4; ++kk) {
      // one contiguous 32B fragment load per matrix (global_load_b128 x2)
      v16h bg = gswz[(j * 4 + kk) * 32 + lane];
      v16h bl = lswz[(j * 4 + kk) * 32 + lane];
      acc_h = __builtin_amdgcn_wmma_f32_16x16x32_f16(false, a[kk], false, bg,
                                                     (short)0, acc_h, false, false);
      acc_l = __builtin_amdgcn_wmma_f32_16x16x32_f16(false, a[kk], false, bl,
                                                     (short)0, acc_l, false, false);
    }
    const float lb = lin_b[col];
    const int rbase = row0 + wave * 16 + half_id * 8;
    if (full) {                            // fast path: unguarded stores
#pragma unroll
      for (int v = 0; v < 8; ++v) {
        size_t o = (size_t)(rbase + v) * D + col;
        h_out [o] = (_Float16)acc_h[v];
        xl_out[o] = acc_l[v] + lb;
      }
    } else {                               // tail block only
#pragma unroll
      for (int v = 0; v < 8; ++v) {
        int gr = rbase + v;
        if (gr < N) {
          h_out [(size_t)gr * D + col] = (_Float16)acc_h[v];
          xl_out[(size_t)gr * D + col] = acc_l[v] + lb;
        }
      }
    }
  }
}

// ---------------- edge scatter: zacc[dst] += h[src] * dinv[s]*dinv[d] -------
__global__ void k_scatter(const int* __restrict__ src, const int* __restrict__ dst,
                          const float* __restrict__ dinv,
                          const _Float16* __restrict__ h,
                          float* __restrict__ zacc, int E) {
  long long gid = (long long)blockIdx.x * blockDim.x + threadIdx.x;
  long long e = gid >> 5;                  // 32 lanes per edge
  if (e >= E) return;
  int c0 = (int)(gid & 31) * 4;            // 4 columns per lane
  int s = src[e], d = dst[e];
  float norm = dinv[s] * dinv[d];
  const _Float16* hp = h + (size_t)s * D + c0;
  float* zp = zacc + (size_t)d * D + c0;
  if (e + 1 < E) __builtin_prefetch(h + (size_t)src[e + 1] * D + c0, 0, 0);
  atomicAdd(zp + 0, (float)hp[0] * norm);
  atomicAdd(zp + 1, (float)hp[1] * norm);
  atomicAdd(zp + 2, (float)hp[2] * norm);
  atomicAdd(zp + 3, (float)hp[3] * norm);
}

// ---------------- combine: zt = tanh(zacc + h*dinv^2 + gcn_b) (in place) ----
__global__ void k_combine(float* __restrict__ zacc, const _Float16* __restrict__ h,
                          const float* __restrict__ dinv,
                          const float* __restrict__ gcn_b, int totalZ) {
  int idx = blockIdx.x * blockDim.x + threadIdx.x;
  if (idx >= totalZ) return;
  int i = idx >> 7, c = idx & (D - 1);
  float di = dinv[i];
  float z = zacc[idx] + (float)h[idx] * di * di + gcn_b[c];
  zacc[idx] = tanhf(z);
}

// -------- fused gate GEMM + epilogue + BN partial sums ----------------------
__global__ __launch_bounds__(GEMM_THREADS)
void k_gate(const float* __restrict__ zt,
            const v16h* __restrict__ wswz,   // gate_w swizzled fragments
            const float* __restrict__ gate_b, const float* __restrict__ xl,
            float* __restrict__ out, float* __restrict__ colsum,
            float* __restrict__ colsq, int N) {
  const int tid  = threadIdx.x;
  const int wave = tid >> 5;
  const int lane = tid & 31;
  const int row0 = blockIdx.x * MT;
  const bool full = (row0 + MT) <= N;

  __shared__ _Float16 sA[MT][D + 8];
  __shared__ float s_sum[D], s_sq[D];
  if (tid < D) { s_sum[tid] = 0.0f; s_sq[tid] = 0.0f; }
  for (int idx = tid; idx < MT * D; idx += GEMM_THREADS) {
    int r = idx >> 7, c = idx & (D - 1);
    int gr = row0 + r;
    float v = (full || gr < N) ? zt[(size_t)gr * D + c] : 0.0f;
    sA[r][c] = (_Float16)v;
  }
  __syncthreads();

  const int half_id = lane >> 4;
  const int mloc    = lane & 15;

  v16h a[4];
#pragma unroll
  for (int kk = 0; kk < 4; ++kk) {
#pragma unroll
    for (int e = 0; e < 16; ++e) {
      int kl = ((e < 8) ? 0 : 16) + half_id * 8 + (e & 7);
      a[kk][e] = sA[wave * 16 + mloc][kk * 32 + kl];
    }
  }

#pragma unroll
  for (int j = 0; j < 8; ++j) {
    v8f acc = {};
    const int col = j * 16 + mloc;
#pragma unroll
    for (int kk = 0; kk < 4; ++kk) {
      v16h bg = wswz[(j * 4 + kk) * 32 + lane];
      acc = __builtin_amdgcn_wmma_f32_16x16x32_f16(false, a[kk], false, bg,
                                                   (short)0, acc, false, false);
    }
    const float gb = gate_b[col];
    const int rbase = row0 + wave * 16 + half_id * 8;
    const int lrow  = wave * 16 + half_id * 8;
    float lsum = 0.0f, lsq = 0.0f;
    if (full) {                            // fast path: unguarded
#pragma unroll
      for (int v = 0; v < 8; ++v) {
        size_t o = (size_t)(rbase + v) * D + col;
        float g  = 1.0f / (1.0f + __expf(-(acc[v] + gb)));
        float zv = (float)sA[lrow + v][col];           // zt row from LDS
        float xv = xl[o];
        float ov = fmaxf((1.0f - g) * xv + g * zv, 0.0f);
        out[o] = ov;
        lsum += ov; lsq += ov * ov;
      }
    } else {
#pragma unroll
      for (int v = 0; v < 8; ++v) {
        int gr = rbase + v;
        if (gr < N) {
          size_t o = (size_t)gr * D + col;
          float g  = 1.0f / (1.0f + __expf(-(acc[v] + gb)));
          float zv = (float)sA[lrow + v][col];
          float xv = xl[o];
          float ov = fmaxf((1.0f - g) * xv + g * zv, 0.0f);
          out[o] = ov;
          lsum += ov; lsq += ov * ov;
        }
      }
    }
    atomicAdd(&s_sum[col], lsum);   // LDS atomics (ds_add_f32)
    atomicAdd(&s_sq[col],  lsq);
  }
  __syncthreads();
  if (tid < D) {
    atomicAdd(&colsum[tid], s_sum[tid]);
    atomicAdd(&colsq[tid],  s_sq[tid]);
  }
}

// ---------------- BN: per-column scale/shift --------------------------------
__global__ void k_bn_stats(const float* __restrict__ colsum,
                           const float* __restrict__ colsq,
                           const float* __restrict__ gamma,
                           const float* __restrict__ beta,
                           float* __restrict__ scale, float* __restrict__ shift,
                           int N) {
  int c = threadIdx.x;
  if (c >= D) return;
  float invN = 1.0f / (float)N;
  float mean = colsum[c] * invN;
  float var  = fmaxf(colsq[c] * invN - mean * mean, 0.0f);   // biased var
  float s = gamma[c] * rsqrtf(var + 1e-5f);
  scale[c] = s;
  shift[c] = beta[c] - mean * s;
}

__global__ void k_bn_apply(float* __restrict__ out, const float* __restrict__ scale,
                           const float* __restrict__ shift, int totalZ) {
  int idx = blockIdx.x * blockDim.x + threadIdx.x;
  if (idx >= totalZ) return;
  int c = idx & (D - 1);
  out[idx] = out[idx] * scale[c] + shift[c];
}

// ---------------------------------------------------------------------------
extern "C" void kernel_launch(void* const* d_in, const int* in_sizes, int n_in,
                              void* d_out, int out_size, void* d_ws, size_t ws_size,
                              hipStream_t stream) {
  const float* x        = (const float*)d_in[0];
  const int*   edge     = (const int*)d_in[1];   // [2, E]: row0=src, row1=dst
  const float* gcn_w    = (const float*)d_in[2];
  const float* gcn_b    = (const float*)d_in[3];
  const float* lin_w    = (const float*)d_in[4];
  const float* lin_b    = (const float*)d_in[5];
  const float* gate_w   = (const float*)d_in[6];
  const float* gate_b   = (const float*)d_in[7];
  const float* bn_gamma = (const float*)d_in[8];
  const float* bn_beta  = (const float*)d_in[9];
  float* out = (float*)d_out;

  const int N = in_sizes[0] / D;
  const int E = in_sizes[1] / 2;
  const int totalZ = N * D;

  // ---- workspace carving (256B aligned) ----
  char* ws = (char*)d_ws;
  size_t off = 0;
  auto carve = [&](size_t bytes) -> void* {
    void* p = ws + off;
    off = (off + bytes + 255) & ~(size_t)255;
    return p;
  };
  float*    deg    = (float*)   carve((size_t)N * 4);          // deg -> dinv
  _Float16* h16    = (_Float16*)carve((size_t)totalZ * 2);     // h (f16)
  float*    xl     = (float*)   carve((size_t)totalZ * 4);     // x@lin_w + b
  float*    zacc   = (float*)   carve((size_t)totalZ * 4);     // scatter acc -> zt
  _Float16* gswz   = (_Float16*)carve((size_t)NFRAG * 16 * 2); // gcn_w fragments
  _Float16* lswz   = (_Float16*)carve((size_t)NFRAG * 16 * 2); // lin_w fragments
  _Float16* tswz   = (_Float16*)carve((size_t)NFRAG * 16 * 2); // gate_w fragments
  float*    colsum = (float*)   carve(D * 4);
  float*    colsq  = (float*)   carve(D * 4);
  float*    scale  = (float*)   carve(D * 4);
  float*    shift  = (float*)   carve(D * 4);
  (void)ws_size;

  const int T = 256;
  k_init     <<<(totalZ + T - 1) / T, T, 0, stream>>>(deg, zacc, colsum, colsq, N, totalZ);
  k_swizzle_w<<<32, 32, 0, stream>>>(gcn_w,  gswz);
  k_swizzle_w<<<32, 32, 0, stream>>>(lin_w,  lswz);
  k_swizzle_w<<<32, 32, 0, stream>>>(gate_w, tswz);
  k_degree   <<<(E + T - 1) / T, T, 0, stream>>>(edge + E, deg, E);
  k_dinv     <<<(N + T - 1) / T, T, 0, stream>>>(deg, N);
  k_gemm_x   <<<(N + MT - 1) / MT, GEMM_THREADS, 0, stream>>>(x, (const v16h*)gswz,
                                                              (const v16h*)lswz,
                                                              lin_b, h16, xl, N);
  long long sth = (long long)E * 32;
  k_scatter  <<<(unsigned)((sth + T - 1) / T), T, 0, stream>>>(edge, edge + E, deg,
                                                               h16, zacc, E);
  k_combine  <<<(totalZ + T - 1) / T, T, 0, stream>>>(zacc, h16, deg, gcn_b, totalZ);
  k_gate     <<<(N + MT - 1) / MT, GEMM_THREADS, 0, stream>>>(zacc, (const v16h*)tswz,
                                                              gate_b, xl, out,
                                                              colsum, colsq, N);
  k_bn_stats <<<1, D, 0, stream>>>(colsum, colsq, bn_gamma, bn_beta, scale, shift, N);
  k_bn_apply <<<(totalZ + T - 1) / T, T, 0, stream>>>(out, scale, shift, totalZ);
}
